// KoLeoLoss_19696720019674
// MI455X (gfx1250) — matrix operand order repaired and checked by reading, not compiled, not run on hardware
//
#include <hip/hip_runtime.h>
#include <hip/hip_bf16.h>
#include <math.h>

// ---------------------------------------------------------------------------
// KoLeo loss, MI455X (gfx1250, wave32, WMMA).
//   N=16384 rows, D=768, fp32 input, scalar fp32 output.
//   Heavy phase: 16384x16384x768 similarity + per-row argmax via
//   v_wmma_f32_16x16x32_f16.  Each WG owns 4 M-tiles (64 rows) staged
//   chunk-major in 96KB LDS; each wave computes a 4M x 2J block of C tiles
//   -> 8 independent WMMA chains, 1 LDS load + 0.5 global loads per WMMA,
//   ~6GB total L2 traffic vs ~82us of WMMA compute.
// ---------------------------------------------------------------------------

#define KN 16384
#define KD 768
#define KEPS 1e-8f
#define NTILES (KN / 16)     // 1024 16x16 tiles per dimension
#define KCHUNKS (KD / 32)    // 24 K-chunks of 32 per WMMA
#define MT 4                 // M-tiles per workgroup
#define ROWS (MT * 16)       // 64 rows per workgroup
#define CHUNKH (ROWS * 32)   // halfs per K-chunk block in LDS (2048)

typedef __attribute__((ext_vector_type(16))) _Float16 v16h;
typedef __attribute__((ext_vector_type(8)))  _Float16 v8h;
typedef __attribute__((ext_vector_type(8)))  float    v8f;

// A fragment: per-lane two contiguous 16B chunks at +0 and +16 halfs.
__device__ __forceinline__ v16h ld_fragA(const _Float16* p) {
  v8h lo = *(const v8h*)p;
  v8h hi = *(const v8h*)(p + 16);
  v16h r;
#pragma unroll
  for (int i = 0; i < 8; ++i) { r[i] = lo[i]; r[i + 8] = hi[i]; }
  return r;
}

// B fragment: per-lane two contiguous 16B chunks at +0 and +16 halfs.
__device__ __forceinline__ v16h ld_fragB(const _Float16* p) {
  v8h lo = *(const v8h*)p;
  v8h hi = *(const v8h*)(p + 16);
  v16h r;
#pragma unroll
  for (int i = 0; i < 8; ++i) { r[i] = lo[i]; r[i + 8] = hi[i]; }
  return r;
}

__device__ __forceinline__ v8f wmma_f16(v16h a, v16h b, v8f c) {
  return __builtin_amdgcn_wmma_f32_16x16x32_f16(
      false, a, false, b, (short)0, c, false, false);
}

// ---------------------------------------------------------------------------
// Kernel 1: row L2-normalize.  One 256-thread block per row.
// ---------------------------------------------------------------------------
__global__ __launch_bounds__(256) void koleo_normalize(
    const float* __restrict__ a, _Float16* __restrict__ xh,
    float* __restrict__ invb) {
  __shared__ float sbuf[256];
  const int row = blockIdx.x;
  const int tid = threadIdx.x;
  const float* ar = a + (size_t)row * KD;

  float e0 = ar[tid];
  float e1 = ar[tid + 256];
  float e2 = ar[tid + 512];
  sbuf[tid] = e0 * e0 + e1 * e1 + e2 * e2;
  __syncthreads();
  for (int s = 128; s > 0; s >>= 1) {
    if (tid < s) sbuf[tid] += sbuf[tid + s];
    __syncthreads();
  }
  const float inv = 1.0f / fmaxf(sqrtf(sbuf[0]), KEPS);

  _Float16* xr = xh + (size_t)row * KD;
  xr[tid]       = (_Float16)(e0 * inv);
  xr[tid + 256] = (_Float16)(e1 * inv);
  xr[tid + 512] = (_Float16)(e2 * inv);
  if (tid == 0) invb[row] = inv;
}

// ---------------------------------------------------------------------------
// Kernel 2: nearest-neighbor argmax over cosine sims via WMMA.
// grid = 256 blocks; each covers 4 M-tiles (64 rows).  A strip stored
// CHUNK-MAJOR in LDS: [kchunk][row][32 halfs] so all 4 M-tile fragments of
// one K-chunk sit within the 16-bit DS offset range of one base register.
// 8 wave32's split the 512 J-tile pairs; each wave computes a 4M x 2J grid
// of C tiles (8 independent WMMA chains).  Argmax reduction scratch (64KB)
// overlays the A buffer after the J loop.
// ---------------------------------------------------------------------------
__global__ __launch_bounds__(256) void koleo_nn_argmax(
    const _Float16* __restrict__ xh, int* __restrict__ ind) {
  __shared__ __align__(16) unsigned char smem[ROWS * KD * 2];  // 96 KB
  _Float16* As = (_Float16*)smem;

  const int tid   = threadIdx.x;
  const int Ibase = blockIdx.x * ROWS;

  // ---- stage A strip chunk-major: LDS[c*2048 + r*32 + p*8] = row r,
  //      K-chunk c, 16B piece p.  6144 16B chunks, 24 per thread. ----
  for (int q = tid; q < ROWS * (KD / 8); q += 256) {
    const int r = q / (KD / 8);
    const int j = q % (KD / 8);        // 16B piece within row (0..95)
    const int c = j >> 2;              // K-chunk (0..23)
    const int p = j & 3;               // piece within chunk (0..3)
    *(v8h*)(As + c * CHUNKH + r * 32 + p * 8) =
        *(const v8h*)(xh + (size_t)(Ibase + r) * KD + j * 8);
  }
  __syncthreads();

  const int  lane = tid & 31;
  const int  wave = tid >> 5;
  const int  nc   = lane & 15;          // N-column / M-row residue
  const bool hi   = lane >= 16;

  // Per-lane A base (hi half-wave owns K offset +8 within the chunk);
  // chunk c adds c*CHUNKH, M-tile t adds t*512 halfs (1024B, DS-immediate).
  const _Float16* ab = As + nc * 32 + (hi ? 8 : 0);

  // Running best per lane: entry t*8+v -> row Ibase + t*16 + v + 8*hi.
  float bestv[MT * 8];
  int   besti[MT * 8];
#pragma unroll
  for (int e = 0; e < MT * 8; ++e) { bestv[e] = -1e30f; besti[e] = 0; }

  for (int p = wave; p < NTILES / 2; p += 8) {
    const int J0 = p * 2;               // tiles J0, J0+1
    const _Float16* b0row =
        xh + (size_t)(J0 * 16 + nc) * KD + (hi ? 8 : 0);
    const _Float16* b1row = b0row + 16 * KD;

    // Prefetch this lane's B rows for the next pair (global_prefetch_b8).
    if (p + 8 < NTILES / 2) {
      __builtin_prefetch(b0row + (size_t)256 * KD, 0, 1);
      __builtin_prefetch(b1row + (size_t)256 * KD, 0, 1);
    }

    v8f acc[MT][2];
#pragma unroll
    for (int t = 0; t < MT; ++t) { acc[t][0] = (v8f){}; acc[t][1] = (v8f){}; }

#pragma unroll 2
    for (int c = 0; c < KCHUNKS; ++c) {
      const _Float16* ac = ab + c * CHUNKH;
      const int k0 = c * 32;
      const v16h B0 = ld_fragB(b0row + k0);
      const v16h B1 = ld_fragB(b1row + k0);
#pragma unroll
      for (int t = 0; t < MT; ++t) {
        const v16h A = ld_fragA(ac + t * 512);
        acc[t][0] = wmma_f16(A, B0, acc[t][0]);
        acc[t][1] = wmma_f16(A, B1, acc[t][1]);
      }
    }

    // C layout: VGPR v -> (row = v + 8*hi, col = Jbase + lane%16).
    const int gcol0 = J0 * 16 + nc;
    const int gcol1 = gcol0 + 16;
#pragma unroll
    for (int t = 0; t < MT; ++t) {
#pragma unroll
      for (int v = 0; v < 8; ++v) {
        const int grow = Ibase + t * 16 + v + (hi ? 8 : 0);
        const int e    = t * 8 + v;
        float x;
        x = (gcol0 == grow) ? -3.0f : acc[t][0][v];
        if (x > bestv[e]) { bestv[e] = x; besti[e] = gcol0; }
        x = (gcol1 == grow) ? -3.0f : acc[t][1][v];
        if (x > bestv[e]) { bestv[e] = x; besti[e] = gcol1; }
      }
    }
  }

  // ---- cross-lane + cross-wave argmax reduction (overlay on As) ----
  __syncthreads();                        // A strip no longer needed
  float* s_bv = (float*)smem;             // [256][32] = 32 KB
  int*   s_bi = (int*)(smem + 32768);     // [256][32] = 32 KB
#pragma unroll
  for (int e = 0; e < MT * 8; ++e) {
    s_bv[tid * 32 + e] = bestv[e];
    s_bi[tid * 32 + e] = besti[e];
  }
  __syncthreads();

  if (tid < ROWS) {
    const int r     = tid;                // local row 0..63
    const int mtile = r >> 4;
    const int rl    = r & 15;
    const int e     = mtile * 8 + (rl & 7);
    const int lsel  = (rl < 8) ? 0 : 16;  // half-wave owning this row
    float bv = -1e30f;
    int   bi = 0;
    for (int w = 0; w < 8; ++w) {
      for (int l = 0; l < 16; ++l) {
        const int t = w * 32 + lsel + l;
        const float cv = s_bv[t * 32 + e];
        if (cv > bv) { bv = cv; bi = s_bi[t * 32 + e]; }
      }
    }
    ind[Ibase + r] = bi;
  }
}

// ---------------------------------------------------------------------------
// Kernel 3: exact fp32 pairwise distance + log per row.
// ---------------------------------------------------------------------------
__global__ __launch_bounds__(256) void koleo_rowdist(
    const float* __restrict__ a, const float* __restrict__ invb,
    const int* __restrict__ ind, float* __restrict__ logd) {
  __shared__ float sbuf[256];
  const int i   = blockIdx.x;
  const int tid = threadIdx.x;
  const int j   = ind[i];
  const float invi = invb[i];
  const float invj = invb[j];
  const float* ai = a + (size_t)i * KD;
  const float* aj = a + (size_t)j * KD;

  float p = 0.0f;
#pragma unroll
  for (int t = 0; t < 3; ++t) {
    const int k = tid + t * 256;
    const float d = ai[k] * invi - aj[k] * invj + KEPS;
    p += d * d;
  }
  sbuf[tid] = p;
  __syncthreads();
  for (int s = 128; s > 0; s >>= 1) {
    if (tid < s) sbuf[tid] += sbuf[tid + s];
    __syncthreads();
  }
  if (tid == 0) logd[i] = logf(sqrtf(sbuf[0]) + KEPS);
}

// ---------------------------------------------------------------------------
// Kernel 4: deterministic reduction of 16384 log-distances -> scalar loss.
// ---------------------------------------------------------------------------
__global__ __launch_bounds__(256) void koleo_finalize(
    const float* __restrict__ logd, float* __restrict__ out) {
  __shared__ float sbuf[256];
  const int tid = threadIdx.x;
  float p = 0.0f;
  for (int t = tid; t < KN; t += 256) p += logd[t];
  sbuf[tid] = p;
  __syncthreads();
  for (int s = 128; s > 0; s >>= 1) {
    if (tid < s) sbuf[tid] += sbuf[tid + s];
    __syncthreads();
  }
  if (tid == 0) out[0] = -(sbuf[0] / (float)KN);
}

// ---------------------------------------------------------------------------
// Launch
// ---------------------------------------------------------------------------
extern "C" void kernel_launch(void* const* d_in, const int* in_sizes, int n_in,
                              void* d_out, int out_size, void* d_ws,
                              size_t ws_size, hipStream_t stream) {
  (void)in_sizes; (void)n_in; (void)out_size; (void)ws_size;
  const float* a = (const float*)d_in[0];
  float* out = (float*)d_out;

  // Workspace layout (all offsets 256B-aligned):
  //   [0)          xh   : N*D f16  = 25165824 B
  //   [+25165824)  invb : N  f32   =    65536 B
  //   [+25231360)  ind  : N  i32   =    65536 B
  //   [+25296896)  logd : N  f32   =    65536 B
  char* ws = (char*)d_ws;
  _Float16* xh   = (_Float16*)ws;
  float*    inv  = (float*)(ws + 25165824);
  int*      ind  = (int*)(ws + 25165824 + 65536);
  float*    logd = (float*)(ws + 25165824 + 131072);

  koleo_normalize<<<KN, 256, 0, stream>>>(a, xh, inv);
  koleo_nn_argmax<<<KN / ROWS, 256, 0, stream>>>(xh, ind);
  koleo_rowdist<<<KN, 256, 0, stream>>>(a, inv, ind, logd);
  koleo_finalize<<<1, 256, 0, stream>>>(logd, out);
}
